// Complex_CA_60309930770499
// MI455X (gfx1250) — matrix operand order repaired and checked by reading, not compiled, not run on hardware
//
#include <hip/hip_runtime.h>

#define BN 16
#define CN 4
#define HH 17
#define WW 17
#define NPIX 289
#define NPIX_PAD 292
#define NTILES 19
#define TPIX (NTILES * 16)   // 304
#define NTHREADS 256
#define NWAVES 8

typedef _Float16 v16h __attribute__((ext_vector_type(16)));
typedef float    v8f  __attribute__((ext_vector_type(8)));

__constant__ float SCENT_K[5][5] = {
    {0.0f, 0.125f, 0.25f, 0.125f, 0.0f},
    {0.125f, 0.25f, 0.5f, 0.25f, 0.125f},
    {0.25f, 0.5f, 1.0f, 0.5f, 0.25f},
    {0.125f, 0.25f, 0.5f, 0.25f, 0.125f},
    {0.0f, 0.125f, 0.25f, 0.125f, 0.0f}};

__global__ __launch_bounds__(NTHREADS) void ca_kernel(
    const float* __restrict__ cell_in, const float* __restrict__ food_in,
    const int* __restrict__ steps_p,
    const float* __restrict__ w1, const float* __restrict__ b1,
    const float* __restrict__ w2, const float* __restrict__ b2,
    float* __restrict__ out) {
  __shared__ float s_cell[CN][NPIX_PAD];
  __shared__ float s_x[CN][NPIX_PAD];
  __shared__ float s_scent[NPIX_PAD];
  __shared__ float s_mask[NPIX_PAD];
  __shared__ _Float16 s_y[TPIX][16];          // 12 valid cols, padded to 16, rows 289..303 zero
  __shared__ _Float16 s_h[NWAVES][16][64];    // per-wave hidden staging
  __shared__ float s_liv;

  const int b    = blockIdx.x;
  const int tid  = threadIdx.x;
  const int lane = tid & 31;
  const int wave = tid >> 5;
  const int hi   = lane >> 4;   // 0 or 1
  const int nl   = lane & 15;

  // ---- load sample state into LDS ----
  for (int p = tid; p < NPIX; p += NTHREADS)
    for (int c = 0; c < CN; ++c)
      s_cell[c][p] = cell_in[(b * CN + c) * NPIX + p];

  // zero-init y staging (pad rows/cols stay zero forever)
  for (int i = tid; i < TPIX * 16; i += NTHREADS)
    ((_Float16*)s_y)[i] = (_Float16)0.0f;

  // scent = 5x5 conv of food, zero pad (food constant -> compute once)
  for (int p = tid; p < NPIX; p += NTHREADS) {
    int r = p / WW, c = p % WW;
    float acc = 0.0f;
    for (int dr = -2; dr <= 2; ++dr)
      for (int dc = -2; dc <= 2; ++dc) {
        int rr = r + dr, cc = c + dc;
        if (rr >= 0 && rr < HH && cc >= 0 && cc < WW)
          acc += SCENT_K[dr + 2][dc + 2] * food_in[b * NPIX + rr * WW + cc];
      }
    s_scent[p] = acc;
  }

  // ---- per-lane constant WMMA B-fragments for weights (f16) ----
  // B layout (32x16, f16): lane = col N (mod 16), elem e -> K = (e&7) + 8*hi + 16*(e>>3)
  v16h bw1[4];    // fc1: out chunks j -> outs 16j..16j+15 ; B[K][N] = W1[16j+N][K]
  float bb1[4];
  v16h bw2[2];    // fc2: K chunks f -> K 32f..32f+31 ; B[K][N] = (N<4)?W2[N][K]:0
  float bb2v;
#pragma unroll
  for (int j = 0; j < 4; ++j) {
    v16h f;
#pragma unroll
    for (int e = 0; e < 16; ++e) {
      int K = (e & 7) + 8 * hi + 16 * (e >> 3);
      float v = (K < 12) ? w1[(16 * j + nl) * 12 + K] : 0.0f;
      f[e] = (_Float16)v;
    }
    bw1[j] = f;
    bb1[j] = b1[16 * j + nl];
  }
#pragma unroll
  for (int f2 = 0; f2 < 2; ++f2) {
    v16h f;
#pragma unroll
    for (int e = 0; e < 16; ++e) {
      int K = (e & 7) + 8 * hi + 16 * (e >> 3) + 32 * f2;
      float v = (nl < 4) ? w2[nl * 64 + K] : 0.0f;
      f[e] = (_Float16)v;
    }
    bw2[f2] = f;
  }
  bb2v = (nl < 4) ? b2[nl] : 0.0f;

  __syncthreads();

  // initial living count (wave 0)
  if (wave == 0) {
    int cnt = 0;
    for (int p = lane; p < NPIX; p += 32)
      if (s_cell[0][p] > 0.1f) cnt++;
    for (int m = 1; m < 32; m <<= 1) cnt += __shfl_xor(cnt, m, 32);
    if (lane == 0) s_liv = (float)cnt;
  }
  __syncthreads();

  const int steps = steps_p[0];
  for (int s = 0; s < steps; ++s) {
    // (1) ch3 = scent; pre-mask = maxpool3(ch0) > 0.1
    for (int p = tid; p < NPIX; p += NTHREADS) {
      s_cell[3][p] = s_scent[p];
      int r = p / WW, c = p % WW;
      float mx = -1e30f;
      for (int dr = -1; dr <= 1; ++dr)
        for (int dc = -1; dc <= 1; ++dc) {
          int rr = r + dr, cc = c + dc;
          if (rr >= 0 && rr < HH && cc >= 0 && cc < WW) {
            float v = s_cell[0][rr * WW + cc];
            mx = v > mx ? v : mx;
          }
        }
      s_mask[p] = (mx > 0.1f) ? 1.0f : 0.0f;
    }
    __syncthreads();

    // (2) build y = [cell, dx(cell), dy(cell)] per pixel -> f16
    for (int p = tid; p < NPIX; p += NTHREADS) {
      int r = p / WW, c = p % WW;
      for (int ch = 0; ch < CN; ++ch) {
        float n[3][3];
        for (int dr = -1; dr <= 1; ++dr)
          for (int dc = -1; dc <= 1; ++dc) {
            int rr = r + dr, cc = c + dc;
            n[dr + 1][dc + 1] = (rr >= 0 && rr < HH && cc >= 0 && cc < WW)
                                    ? s_cell[ch][rr * WW + cc] : 0.0f;
          }
        float dxv = ((n[0][2] - n[0][0]) + 2.0f * (n[1][2] - n[1][0]) + (n[2][2] - n[2][0])) * 0.125f;
        float dyv = ((n[2][0] - n[0][0]) + 2.0f * (n[2][1] - n[0][1]) + (n[2][2] - n[0][2])) * 0.125f;
        s_y[p][ch]     = (_Float16)n[1][1];
        s_y[p][4 + ch] = (_Float16)dxv;
        s_y[p][8 + ch] = (_Float16)dyv;
      }
    }
    __syncthreads();

    // (3) per-pixel MLP via WMMA; one 16-pixel tile per wave iteration
    for (int t = wave; t < NTILES; t += NWAVES) {
      // A fragment layer1 (16x32 f16): lane row M=nl, elems 0..7 -> K = 8*hi + i, rest 0
      v16h a1;
#pragma unroll
      for (int i = 0; i < 16; ++i) a1[i] = (_Float16)0.0f;
      {
        int pix = t * 16 + nl;
#pragma unroll
        for (int i = 0; i < 8; ++i) a1[i] = s_y[pix][8 * hi + i];
      }
      // layer 1: 4 output chunks of 16, bias in accumulator, ReLU on store
#pragma unroll
      for (int j = 0; j < 4; ++j) {
        v8f c1;
#pragma unroll
        for (int g = 0; g < 8; ++g) c1[g] = bb1[j];
        c1 = __builtin_amdgcn_wmma_f32_16x16x32_f16(false, a1, false, bw1[j],
                                                    (short)0, c1, false, false);
#pragma unroll
        for (int g = 0; g < 8; ++g) {
          float v = c1[g];
          v = v > 0.0f ? v : 0.0f;
          s_h[wave][g + 8 * hi][16 * j + nl] = (_Float16)v;   // row = pixel-in-tile
        }
      }
      // layer 2: K=64 as two chained K=32 WMMAs, bias in accumulator
      v8f c2;
#pragma unroll
      for (int g = 0; g < 8; ++g) c2[g] = bb2v;
#pragma unroll
      for (int f2 = 0; f2 < 2; ++f2) {
        v16h a2;
#pragma unroll
        for (int i = 0; i < 8; ++i) {
          a2[i]     = s_h[wave][nl][32 * f2 + 8 * hi + i];
          a2[8 + i] = s_h[wave][nl][32 * f2 + 16 + 8 * hi + i];
        }
        c2 = __builtin_amdgcn_wmma_f32_16x16x32_f16(false, a2, false, bw2[f2],
                                                    (short)0, c2, false, false);
      }
      // x = cell + delta   (D layout: VGPR g -> pixel row g+8*hi, col nl = channel)
      if (nl < 4) {
#pragma unroll
        for (int g = 0; g < 8; ++g) {
          int p = t * 16 + g + 8 * hi;
          if (p < NPIX) s_x[nl][p] = s_cell[nl][p] + c2[g];
        }
      }
    }
    __syncthreads();

    // (4) post-mask from maxpool3(x0), combine with pre-mask, clip -> s_cell
    for (int p = tid; p < NPIX; p += NTHREADS) {
      int r = p / WW, c = p % WW;
      float mx = -1e30f;
      for (int dr = -1; dr <= 1; ++dr)
        for (int dc = -1; dc <= 1; ++dc) {
          int rr = r + dr, cc = c + dc;
          if (rr >= 0 && rr < HH && cc >= 0 && cc < WW) {
            float v = s_x[0][rr * WW + cc];
            mx = v > mx ? v : mx;
          }
        }
      bool m = (s_mask[p] != 0.0f) && (mx > 0.1f);
      for (int ch = 0; ch < CN; ++ch) {
        float v = m ? s_x[ch][p] : 0.0f;
        v = fminf(10.0f, fmaxf(-10.0f, v));
        if (ch == 0) v = fminf(1.0f, fmaxf(0.0f, v));
        s_cell[ch][p] = v;
      }
    }
    __syncthreads();

    // (5) keep-k-largest on ch0 (k = previous living), then new living (wave 0 only)
    if (wave == 0) {
      float vals[10];
      unsigned bits[10];
      int np = 0;
      for (int p = lane; p < NPIX; p += 32) {
        float v = s_cell[0][p];
        vals[np] = v;
        bits[np] = __float_as_uint(v);
        np++;
      }
      int k = (int)s_liv - 1;
      if (k < 0) k += NPIX;
      k += 1;  // 1-indexed rank of threshold
      // values in [0,1]: bit pattern is monotone; bisect for largest u with count_ge(u) >= k
      unsigned lo = 0u, hb = 0x3F800001u;
      while (hb - lo > 1u) {
        unsigned mid = (lo + hb) >> 1;
        int cnt = 0;
        for (int i = 0; i < np; ++i) cnt += (bits[i] >= mid) ? 1 : 0;
        for (int m = 1; m < 32; m <<= 1) cnt += __shfl_xor(cnt, m, 32);
        if (cnt >= k) lo = mid; else hb = mid;
      }
      float kth = __uint_as_float(lo);
      int cnt2 = 0, idx = 0;
      for (int p = lane; p < NPIX; p += 32) {
        float v = vals[idx++];
        float mv = (v >= kth) ? v : 0.0f;
        s_cell[0][p] = mv;
        cnt2 += (mv > 0.1f) ? 1 : 0;
      }
      for (int m = 1; m < 32; m <<= 1) cnt2 += __shfl_xor(cnt2, m, 32);
      if (lane == 0) s_liv = (float)cnt2;
    }
    __syncthreads();
  }

  // ---- write outputs: cell, food, total_pixel_val, living_count ----
  for (int p = tid; p < NPIX; p += NTHREADS) {
    for (int c = 0; c < CN; ++c)
      out[(b * CN + c) * NPIX + p] = s_cell[c][p];
    out[BN * CN * NPIX + b * NPIX + p] = food_in[b * NPIX + p];
  }
  if (wave == 0) {
    float ssum = 0.0f;
    for (int p = lane; p < NPIX; p += 32) ssum += s_cell[0][p];
    for (int m = 1; m < 32; m <<= 1) ssum += __shfl_xor(ssum, m, 32);
    if (lane == 0) {
      out[BN * CN * NPIX + BN * NPIX + b]      = ssum;
      out[BN * CN * NPIX + BN * NPIX + BN + b] = s_liv;
    }
  }
}

extern "C" void kernel_launch(void* const* d_in, const int* in_sizes, int n_in,
                              void* d_out, int out_size, void* d_ws, size_t ws_size,
                              hipStream_t stream) {
  (void)in_sizes; (void)n_in; (void)out_size; (void)d_ws; (void)ws_size;
  const float* cell  = (const float*)d_in[0];
  const float* food  = (const float*)d_in[1];
  const int*   steps = (const int*)d_in[2];
  const float* w1    = (const float*)d_in[3];
  const float* b1    = (const float*)d_in[4];
  const float* w2    = (const float*)d_in[5];
  const float* b2    = (const float*)d_in[6];
  ca_kernel<<<BN, NTHREADS, 0, stream>>>(cell, food, steps, w1, b1, w2, b2,
                                         (float*)d_out);
}
